// StackedLSTM_45526653337956
// MI455X (gfx1250) — compile-verified, matrix operand order checked by reading, not run on hardware
//
#include <hip/hip_runtime.h>

// ---------------------------------------------------------------------------
// Stacked LSTM for MI455X (gfx1250, wave32, WMMA).
//   B=64, T=512, H=1024, L=4.
//   Phase A (per layer): x_proj = X * Wih^T + (b_ih+b_hh)  -- big bf16 WMMA GEMM
//   Phase B (per layer): persistent scan, 32 blocks x 8 waves. Each block owns
//     2 hidden-tiles x 4 batch-tiles; its Whh working set (256KB bf16) is staged
//     into LDS once per layer (bank-conflict-free padded layout), so the 512
//     serial timesteps read weights via ds_load_b128 instead of L2. Cell state
//     c stays in 8 VGPRs across all timesteps; h is double-buffered in global
//     bf16 with one grid barrier per step; next step's x_proj gate tiles are
//     prefetched (global_prefetch_b8) during the current step's WMMA K-loop.
// ---------------------------------------------------------------------------

#define B_ 64
#define T_ 512
#define H_ 1024
#define L_ 4
#define G_ 4096   // 4*H
#define LDP 1032  // padded LDS row stride (bf16 elems): 1024+8 -> 4-bank row skew

typedef __bf16 bf16_t;
typedef __attribute__((ext_vector_type(16))) __bf16 v16bf;
typedef __attribute__((ext_vector_type(8)))  __bf16 v8bf;
typedef __attribute__((ext_vector_type(8)))  float  v8f;

static __device__ __forceinline__ bf16_t to_bf16(float f) {
  unsigned u = __builtin_bit_cast(unsigned, f);
  unsigned r = (u + 0x7FFFu + ((u >> 16) & 1u)) >> 16; // RNE
  unsigned short s = (unsigned short)r;
  return __builtin_bit_cast(bf16_t, s);
}

// 16-bit A(16x32)/B(32x16 from row-major [N,K]) fragment, CDNA5 layout:
// lanes 0-15: row r, K=[0..7],[16..23]; lanes 16-31: row r, K=[8..15],[24..31].
static __device__ __forceinline__ v16bf load_frag(const bf16_t* p, int ld) {
  unsigned lane = threadIdx.x & 31u;
  unsigned r    = lane & 15u;
  unsigned hi   = lane >> 4;
  const bf16_t* q = p + (size_t)r * (size_t)ld + hi * 8u;
  v8bf lo = *(const v8bf*)(q);
  v8bf hh = *(const v8bf*)(q + 16);
  v16bf out;
#pragma unroll
  for (int i = 0; i < 8; ++i) { out[i] = lo[i]; out[i + 8] = hh[i]; }
  return out;
}

static __device__ __forceinline__ v8f wmma_bf16(v16bf a, v16bf b, v8f c) {
  return __builtin_amdgcn_wmma_f32_16x16x32_bf16(false, a, false, b, (short)0, c,
                                                 false, false);
}

#define V8F_ZERO {0.f,0.f,0.f,0.f,0.f,0.f,0.f,0.f}

// monotonic ticket grid barrier (bar must be 0 before first use in a launch)
static __device__ __forceinline__ void grid_sync(unsigned* bar, unsigned nblk) {
  __threadfence();
  __syncthreads();
  if (threadIdx.x == 0) {
    unsigned ticket = __hip_atomic_fetch_add(bar, 1u, __ATOMIC_ACQ_REL,
                                             __HIP_MEMORY_SCOPE_AGENT);
    unsigned target = (ticket / nblk + 1u) * nblk;
    while (__hip_atomic_load(bar, __ATOMIC_ACQUIRE, __HIP_MEMORY_SCOPE_AGENT) < target) {
      __builtin_amdgcn_s_sleep(1);
    }
    __threadfence();
  }
  __syncthreads();
}

// ---------------------------------------------------------------------------
// Prep kernels
// ---------------------------------------------------------------------------
__global__ void prep_weights(const float* __restrict__ Wih, const float* __restrict__ Whh,
                             const float* __restrict__ bih, const float* __restrict__ bhh,
                             bf16_t* __restrict__ wih_b, bf16_t* __restrict__ whh_b,
                             float* __restrict__ bias) {
  const size_t nw = (size_t)L_ * G_ * H_;
  const size_t nb = (size_t)L_ * G_;
  size_t i = (size_t)blockIdx.x * blockDim.x + threadIdx.x;
  size_t stride = (size_t)gridDim.x * blockDim.x;
  for (size_t k = i; k < nw; k += stride) {
    wih_b[k] = to_bf16(Wih[k]);
    whh_b[k] = to_bf16(Whh[k]);
  }
  for (size_t k = i; k < nb; k += stride) bias[k] = bih[k] + bhh[k];
}

// x [B,T,H] f32  ->  xb [T,B,H] bf16 (time-major for the scan)
__global__ void prep_x(const float* __restrict__ x, bf16_t* __restrict__ xb) {
  const size_t n = (size_t)T_ * B_ * H_;
  size_t i = (size_t)blockIdx.x * blockDim.x + threadIdx.x;
  size_t stride = (size_t)gridDim.x * blockDim.x;
  for (size_t k = i; k < n; k += stride) {
    size_t h = k % H_;
    size_t b = (k / H_) % B_;
    size_t t = k / ((size_t)H_ * B_);
    xb[k] = to_bf16(x[((size_t)b * T_ + t) * H_ + h]);
  }
}

// ---------------------------------------------------------------------------
// Phase A: x_proj[M=T*B, 4H] = A[M,H] * W[4H,H]^T + bias   (bf16 WMMA, f32 acc)
// grid (M/64, 4H/128), 256 threads = 8 waves (2x4), each wave a 32x32 tile.
// ---------------------------------------------------------------------------
__global__ __launch_bounds__(256) void gemm_xproj(
    const bf16_t* __restrict__ A, const bf16_t* __restrict__ W,
    const float* __restrict__ bias, float* __restrict__ C) {
  const int K = H_, N = G_;
  unsigned w = threadIdx.x >> 5;
  unsigned wm = w & 1u, wn = w >> 1;
  size_t m0 = (size_t)blockIdx.x * 64 + wm * 32;
  size_t n0 = (size_t)blockIdx.y * 128 + wn * 32;

  v8f a00 = V8F_ZERO, a01 = V8F_ZERO, a10 = V8F_ZERO, a11 = V8F_ZERO;

  const bf16_t* Ap = A + m0 * K;
  const bf16_t* Wp = W + n0 * K;
  for (int k0 = 0; k0 < K; k0 += 32) {
    __builtin_prefetch(Ap + k0 + 256, 0, 1);           // global_prefetch_b8
    __builtin_prefetch(Wp + k0 + 256, 0, 1);
    v16bf fa0 = load_frag(Ap + k0, K);
    v16bf fa1 = load_frag(Ap + 16 * K + k0, K);
    v16bf fb0 = load_frag(Wp + k0, K);
    v16bf fb1 = load_frag(Wp + 16 * K + k0, K);
    a00 = wmma_bf16(fa0, fb0, a00);
    a01 = wmma_bf16(fa0, fb1, a01);
    a10 = wmma_bf16(fa1, fb0, a10);
    a11 = wmma_bf16(fa1, fb1, a11);
  }

  unsigned lane = threadIdx.x & 31u;
  unsigned col = lane & 15u, hi = lane >> 4;
#pragma unroll
  for (int v = 0; v < 8; ++v) {
    unsigned row = v + 8 * hi;
    float b0 = bias[n0 + col];
    float b1 = bias[n0 + 16 + col];
    C[(m0 + row) * N + n0 + col]           = a00[v] + b0;
    C[(m0 + row) * N + n0 + 16 + col]      = a01[v] + b1;
    C[(m0 + 16 + row) * N + n0 + col]      = a10[v] + b0;
    C[(m0 + 16 + row) * N + n0 + 16 + col] = a11[v] + b1;
  }
}

// ---------------------------------------------------------------------------
// Phase B: persistent recurrent scan. 32 blocks x 256 threads = 256 waves.
// Block b owns j-tiles {2b, 2b+1}; wave w: mt = w&3 (batch tile), jt2 = w>>2.
// Whh slice for both j-tiles (all 4 gates) is staged in LDS: row r maps to
// (jt2 = r>>6, gate = (r>>4)&3, rr = r&15), padded stride LDP for conflict-free
// 16B fragment reads.
// ---------------------------------------------------------------------------
__global__ __launch_bounds__(256) void lstm_scan(
    const float* __restrict__ xproj,   // [T, B, 4H] f32 (bias included)
    const bf16_t* __restrict__ Whh,    // [4H, H] bf16, layer slice
    bf16_t* __restrict__ hbuf,         // [2, B, H] bf16 double buffer (zeroed)
    bf16_t* __restrict__ ys,           // [T, B, H] bf16 next-layer input
    float* __restrict__ out,           // d_out base (output + h_n + c_n)
    int is_last, int layer, unsigned* __restrict__ bar) {
  extern __shared__ bf16_t sWhh[];     // [128][LDP] = 258 KB

  unsigned tid = threadIdx.x;
  unsigned lane = tid & 31u;
  unsigned w = tid >> 5;               // wave in block: 0..7
  unsigned mt = w & 3u;                // batch tile 0..3
  unsigned jt2 = w >> 2;               // 0..1
  unsigned jt = blockIdx.x * 2u + jt2; // hidden tile 0..63
  unsigned col = lane & 15u, hi = lane >> 4;
  size_t m0 = (size_t)mt * 16;
  size_t j0 = (size_t)jt * 16;

  // --- stage this block's Whh working set into LDS (once per layer) ---
  for (unsigned c = tid; c < 128u * 128u; c += 256u) {
    unsigned r = c >> 7;                 // LDS row 0..127
    unsigned col8 = (c & 127u) * 8u;     // bf16 column, step 8 (16B chunks)
    unsigned sj = r >> 6;                // which j-tile
    unsigned g = (r >> 4) & 3u;          // gate
    unsigned rr = r & 15u;               // row within tile
    size_t grow = (size_t)g * H_ + ((size_t)blockIdx.x * 2u + sj) * 16u + rr;
    v8bf vv = *(const v8bf*)(Whh + grow * H_ + col8);
    *(v8bf*)(&sWhh[(size_t)r * LDP + col8]) = vv;   // ds_store_b128
  }
  __syncthreads();

  const bf16_t* sB = &sWhh[(size_t)jt2 * 64u * LDP]; // this wave's 4 gate tiles

  const size_t OUT_HN = (size_t)B_ * T_ * H_;
  const size_t OUT_CN = OUT_HN + (size_t)L_ * B_ * H_;

  v8f c = V8F_ZERO;

  for (int t = 0; t < T_; ++t) {
    const bf16_t* hread = hbuf + (size_t)(t & 1) * (B_ * H_);
    bf16_t* hwrite      = hbuf + (size_t)((t + 1) & 1) * (B_ * H_);

    // Prefetch next step's x_proj gate tiles (HBM stream) so their latency is
    // hidden under this step's 128 WMMAs. 32 lanes x 2 prefetches cover all
    // 16 rows x 4 gate tiles of this wave's next-step operand.
    if (t + 1 < T_) {
      const float* xpn = xproj + ((size_t)(t + 1) * B_ + m0) * G_ + j0;
      unsigned prow = lane & 15u;
      unsigned pg = lane >> 4;  // 0 or 1
      __builtin_prefetch(xpn + (size_t)prow * G_ + (size_t)pg * H_, 0, 1);
      __builtin_prefetch(xpn + (size_t)prow * G_ + (size_t)(pg + 2) * H_, 0, 1);
    }

    v8f ai = V8F_ZERO, af = V8F_ZERO, ag = V8F_ZERO, ao = V8F_ZERO;
    const bf16_t* abase = hread + m0 * H_;
    for (int k0 = 0; k0 < H_; k0 += 32) {
      v16bf a = load_frag(abase + k0, H_);             // global (L2-hot h)
      v16bf bi = load_frag(sB + (size_t)(0 * 16) * LDP + k0, LDP); // ds_load_b128
      ai = wmma_bf16(a, bi, ai);
      v16bf bf = load_frag(sB + (size_t)(1 * 16) * LDP + k0, LDP);
      af = wmma_bf16(a, bf, af);
      v16bf bg = load_frag(sB + (size_t)(2 * 16) * LDP + k0, LDP);
      ag = wmma_bf16(a, bg, ag);
      v16bf bo = load_frag(sB + (size_t)(3 * 16) * LDP + k0, LDP);
      ao = wmma_bf16(a, bo, ao);
    }

    const float* xp = xproj + ((size_t)t * B_ + m0) * G_ + j0;
#pragma unroll
    for (int v = 0; v < 8; ++v) {
      unsigned row = v + 8 * hi;
      size_t ro = (size_t)row * G_ + col;
      float gi = ai[v] + xp[ro];
      float gf = af[v] + xp[ro + H_];
      float gg = ag[v] + xp[ro + 2 * H_];
      float go = ao[v] + xp[ro + 3 * H_];
      float I = 1.f / (1.f + __expf(-gi));
      float F = 1.f / (1.f + __expf(-gf));
      float Gg = tanhf(gg);
      float O = 1.f / (1.f + __expf(-go));
      float cn = F * c[v] + I * Gg;
      c[v] = cn;
      float h = O * tanhf(cn);

      size_t b = m0 + row;
      size_t j = j0 + col;
      hwrite[b * H_ + j] = to_bf16(h);
      if (is_last) {
        out[(b * T_ + (size_t)t) * H_ + j] = h;   // output [B,T,H]
      } else {
        ys[((size_t)t * B_ + b) * H_ + j] = to_bf16(h);
      }
      if (t == T_ - 1) {
        out[OUT_HN + (size_t)layer * B_ * H_ + b * H_ + j] = h;
        out[OUT_CN + (size_t)layer * B_ * H_ + b * H_ + j] = c[v];
      }
    }
    grid_sync(bar, gridDim.x);
  }
}

// ---------------------------------------------------------------------------
extern "C" void kernel_launch(void* const* d_in, const int* in_sizes, int n_in,
                              void* d_out, int out_size, void* d_ws, size_t ws_size,
                              hipStream_t stream) {
  (void)in_sizes; (void)n_in; (void)out_size; (void)ws_size;
  const float* x   = (const float*)d_in[0];
  const float* Wih = (const float*)d_in[1];
  const float* Whh = (const float*)d_in[2];
  const float* bih = (const float*)d_in[3];
  const float* bhh = (const float*)d_in[4];
  float* out = (float*)d_out;

  // --- workspace carve (256B aligned) ---
  char* ws = (char*)d_ws;
  size_t off = 0;
  auto carve = [&](size_t bytes) {
    char* p = ws + off;
    off += (bytes + 255) & ~(size_t)255;
    return p;
  };
  unsigned* bar   = (unsigned*)carve(256);
  bf16_t* wih_b   = (bf16_t*)carve((size_t)L_ * G_ * H_ * sizeof(bf16_t)); // 32MB
  bf16_t* whh_b   = (bf16_t*)carve((size_t)L_ * G_ * H_ * sizeof(bf16_t)); // 32MB
  float*  bias    = (float*) carve((size_t)L_ * G_ * sizeof(float));
  bf16_t* hbuf    = (bf16_t*)carve((size_t)2 * B_ * H_ * sizeof(bf16_t));
  bf16_t* xb0     = (bf16_t*)carve((size_t)T_ * B_ * H_ * sizeof(bf16_t)); // 64MB
  bf16_t* xb1     = (bf16_t*)carve((size_t)T_ * B_ * H_ * sizeof(bf16_t)); // 64MB
  float*  xproj   = (float*) carve((size_t)T_ * B_ * G_ * sizeof(float));  // 512MB

  hipMemsetAsync(bar, 0, 256, stream);

  prep_weights<<<4096, 256, 0, stream>>>(Wih, Whh, bih, bhh, wih_b, whh_b, bias);
  prep_x<<<8192, 256, 0, stream>>>(x, xb0);

  const size_t scan_lds = (size_t)128 * LDP * sizeof(bf16_t); // 258 KB of 320 KB

  bf16_t* xcur = xb0;
  bf16_t* xnext = xb1;
  for (int l = 0; l < L_; ++l) {
    hipMemsetAsync(hbuf, 0, (size_t)2 * B_ * H_ * sizeof(bf16_t), stream);
    gemm_xproj<<<dim3((T_ * B_) / 64, G_ / 128), 256, 0, stream>>>(
        xcur, wih_b + (size_t)l * G_ * H_, bias + (size_t)l * G_, xproj);
    lstm_scan<<<32, 256, scan_lds, stream>>>(
        xproj, whh_b + (size_t)l * G_ * H_, hbuf, xnext, out,
        (l == L_ - 1) ? 1 : 0, l, bar);
    bf16_t* tmp = xcur; xcur = xnext; xnext = tmp;
  }
}
